// GATModelBaseline_12962211299659
// MI455X (gfx1250) — matrix operand order repaired and checked by reading, not compiled, not run on hardware
//
#include <hip/hip_runtime.h>
#include <math.h>

#define N_NODES 50000
#define N_EDGES 800000
#define EPRIME  (N_EDGES + N_NODES)   // edges + self-loops
#define IN_F    256
#define H_HEADS 8
#define C_CH    32
#define HC      256                    // H*C
#define OUT_C   40
#define NEG_SLOPE 0.2f

typedef __attribute__((ext_vector_type(16))) _Float16 v16h;
typedef __attribute__((ext_vector_type(8)))  float    v8f;

// ---------------------------------------------------------------- utilities
__global__ void fill_kernel(float* __restrict__ p, float v, int n) {
    int i = blockIdx.x * blockDim.x + threadIdx.x;
    if (i < n) p[i] = v;
}

__device__ __forceinline__ void edge_nodes(const int* __restrict__ ei, int e,
                                           int& sn, int& dn) {
    if (e < N_EDGES) { sn = ei[e]; dn = ei[N_EDGES + e]; }
    else             { sn = e - N_EDGES; dn = sn; }      // self loop
}

// float atomic max via monotone int mapping (init with -INFINITY)
__device__ __forceinline__ void atomicMaxF(float* addr, float val) {
    if (val >= 0.0f) atomicMax((int*)addr, __float_as_int(val));
    else             atomicMin((unsigned int*)addr, __float_as_uint(val));
}

// ---------------------------------------------------------------- packing
// Pre-swizzle operands into per-lane WMMA fragments (f16), so the GEMM hot
// loop is pure contiguous b128 loads + v_wmma.
// A fragment (16-bit A 16x32, ISA 7.12.2): lane = {half,mr}; element 2v(+1)
//   = A[tm*16+mr][ks*32 + kk], kk = (v<4 ? 2v : 16+2(v-4)) + 8*half.
__global__ void pack_a_kernel(const float* __restrict__ A, v16h* __restrict__ Ap,
                              int M, int K) {
    int i = blockIdx.x * blockDim.x + threadIdx.x;
    int ksteps = K >> 5;
    int total = (M >> 4) * ksteps * 32;
    if (i >= total) return;
    int lane = i & 31;
    int rest = i >> 5;
    int ks = rest % ksteps, tm = rest / ksteps;
    int half = lane >> 4, mr = lane & 15;
    const float* Arow = A + (size_t)(tm * 16 + mr) * K + ks * 32;
    v16h a;
#pragma unroll
    for (int v = 0; v < 8; ++v) {
        int kk = (v < 4 ? 2 * v : 16 + 2 * (v - 4)) + half * 8;
        a[2 * v]     = (_Float16)Arow[kk];
        a[2 * v + 1] = (_Float16)Arow[kk + 1];
    }
    Ap[i] = a;
}

// B fragment (16-bit B 32x16): lane = {half, col mr}; element 2v(+1)
//   = B[ks*32 + kk][tn*16 + mr], kk = (v<4 ? 2v : 16+2(v-4)) + 8*half.
__global__ void pack_b_kernel(const float* __restrict__ B, v16h* __restrict__ Bp,
                              int Nn, int K) {
    int i = blockIdx.x * blockDim.x + threadIdx.x;
    int tilesN = Nn >> 4;
    int total = (K >> 5) * tilesN * 32;
    if (i >= total) return;
    int lane = i & 31;
    int rest = i >> 5;
    int tn = rest % tilesN, ks = rest / tilesN;
    int half = lane >> 4, mr = lane & 15;
    const float* Bcol = B + (size_t)ks * 32 * Nn + tn * 16 + mr;
    v16h b;
#pragma unroll
    for (int v = 0; v < 8; ++v) {
        int kk = (v < 4 ? 2 * v : 16 + 2 * (v - 4)) + half * 8;
        b[2 * v]     = (_Float16)Bcol[(size_t)kk * Nn];
        b[2 * v + 1] = (_Float16)Bcol[(size_t)(kk + 1) * Nn];
    }
    Bp[i] = b;
}

// ---------------------------------------------------------------- WMMA GEMM
// C[M,Nn] = A[M,K] @ B[K,Nn] from pre-packed f16 fragments, f32 accumulate.
// One wave computes a 16x64 strip (4 tiles): one A-fragment load feeds
// 4 back-to-back v_wmma_f32_16x16x32_f16 per K step.
// Requires M%16==0, Nn%64==0, K%32==0 (true for all calls here).
__global__ __launch_bounds__(256)
void gemm_wmma_packed(const v16h* __restrict__ Ap, const v16h* __restrict__ Bp,
                      float* __restrict__ C, int M, int Nn, int K) {
    int wave    = blockIdx.x * (blockDim.x >> 5) + (threadIdx.x >> 5);
    int groupsN = Nn >> 6;                 // 64-column groups
    int total   = (M >> 4) * groupsN;
    if (wave >= total) return;
    int tm = wave / groupsN, tg = wave % groupsN;
    int lane = threadIdx.x & 31;
    int half = lane >> 4, mr = lane & 15;
    int ksteps = K >> 5, tilesN = Nn >> 4;

    v8f accs[4] = {v8f{}, v8f{}, v8f{}, v8f{}};
    const v16h* ApW = Ap + (size_t)tm * ksteps * 32 + lane;
    const v16h* BpW = Bp + (size_t)(tg * 4) * 32 + lane;

    for (int ks = 0; ks < ksteps; ++ks) {
        v16h a = ApW[(size_t)ks * 32];
#pragma unroll
        for (int g = 0; g < 4; ++g) {
            v16h b = BpW[((size_t)ks * tilesN + g) * 32];
            accs[g] = __builtin_amdgcn_wmma_f32_16x16x32_f16(
                false, a, false, b, (short)0, accs[g], false, false);
        }
    }
    // D layout: VGPR r -> row r + 8*half, lane -> col
#pragma unroll
    for (int g = 0; g < 4; ++g) {
        float* Cp = C + (size_t)(tm * 16 + half * 8) * Nn + tg * 64 + g * 16 + mr;
#pragma unroll
        for (int r = 0; r < 8; ++r) Cp[(size_t)r * Nn] = accs[g][r];
    }
}

// ------------------------------------------------------------ GAT kernels
// as[n,h] = <h[n,h,:], a_src[h,:]>, ad likewise. One thread per (n,head).
__global__ void attn_coef_kernel(const float* __restrict__ h,
                                 const float* __restrict__ a_src,
                                 const float* __restrict__ a_dst,
                                 float* __restrict__ as, float* __restrict__ ad) {
    int i = blockIdx.x * blockDim.x + threadIdx.x;
    if (i >= N_NODES * H_HEADS) return;
    int n = i >> 3, hd = i & 7;
    const float* hp = h + (size_t)n * HC + hd * C_CH;
    const float* ap = a_src + hd * C_CH;
    const float* bp = a_dst + hd * C_CH;
    float s1 = 0.f, s2 = 0.f;
#pragma unroll
    for (int c = 0; c < C_CH; ++c) { float v = hp[c]; s1 += v * ap[c]; s2 += v * bp[c]; }
    as[i] = s1; ad[i] = s2;
}

// pass 1: segment max of leaky_relu(e) over dst
__global__ void edge_max_kernel(const int* __restrict__ ei,
                                const float* __restrict__ as,
                                const float* __restrict__ ad,
                                float* __restrict__ m) {
    int i = blockIdx.x * blockDim.x + threadIdx.x;
    if (i >= EPRIME * H_HEADS) return;
    int e = i >> 3, hd = i & 7;
    int sn, dn; edge_nodes(ei, e, sn, dn);
    float v = as[sn * H_HEADS + hd] + ad[dn * H_HEADS + hd];
    v = v > 0.f ? v : NEG_SLOPE * v;
    atomicMaxF(&m[dn * H_HEADS + hd], v);
}

// pass 2: ex = exp(e - m[dst]); segment sum
__global__ void edge_exp_kernel(const int* __restrict__ ei,
                                const float* __restrict__ as,
                                const float* __restrict__ ad,
                                const float* __restrict__ m,
                                float* __restrict__ ex, float* __restrict__ ssum) {
    int i = blockIdx.x * blockDim.x + threadIdx.x;
    if (i >= EPRIME * H_HEADS) return;
    int e = i >> 3, hd = i & 7;
    int sn, dn; edge_nodes(ei, e, sn, dn);
    float v = as[sn * H_HEADS + hd] + ad[dn * H_HEADS + hd];
    v = v > 0.f ? v : NEG_SLOPE * v;
    float exv = expf(v - m[dn * H_HEADS + hd]);
    ex[i] = exv;
    atomicAdd(&ssum[dn * H_HEADS + hd], exv);
}

// pass 3: acc[dst,c] += (1/H) * sum_h alpha[e,h] * h[src,h,c]
// 32 lanes per edge (lane = channel); head-mean folded in -> N x 32 accumulator.
__global__ void edge_msg_kernel(const int* __restrict__ ei,
                                const float* __restrict__ ex,
                                const float* __restrict__ ssum,
                                const float* __restrict__ h,
                                float* __restrict__ acc) {
    int gid = blockIdx.x * blockDim.x + threadIdx.x;
    int e = gid >> 5;
    if (e >= EPRIME) return;
    int c = gid & 31;
    int sn, dn; edge_nodes(ei, e, sn, dn);
    float sum = 0.f;
#pragma unroll
    for (int hd = 0; hd < H_HEADS; ++hd) {
        float alpha = ex[(size_t)e * H_HEADS + hd] / ssum[dn * H_HEADS + hd];
        sum += alpha * h[(size_t)sn * HC + hd * C_CH + c];
    }
    atomicAdd(&acc[dn * C_CH + c], sum * (1.0f / H_HEADS));
}

__global__ void bias_elu_kernel(const float* __restrict__ acc,
                                const float* __restrict__ b,
                                float* __restrict__ t, int n) {
    int i = blockIdx.x * blockDim.x + threadIdx.x;
    if (i >= n) return;
    float v = acc[i] + b[i & (C_CH - 1)];
    t[i] = v > 0.f ? v : (expf(v) - 1.f);
}

__global__ void out_proj_kernel(const float* __restrict__ t,
                                const float* __restrict__ Wo,
                                const float* __restrict__ bo,
                                float* __restrict__ out) {
    int i = blockIdx.x * blockDim.x + threadIdx.x;
    if (i >= N_NODES * OUT_C) return;
    int n = i / OUT_C, o = i % OUT_C;
    const float* tp = t + (size_t)n * C_CH;
    float s = bo[o];
#pragma unroll
    for (int c = 0; c < C_CH; ++c) s += tp[c] * Wo[c * OUT_C + o];
    out[i] = s;
}

__global__ void log_softmax_kernel(float* __restrict__ out) {
    int n = blockIdx.x * blockDim.x + threadIdx.x;
    if (n >= N_NODES) return;
    float* row = out + (size_t)n * OUT_C;
    float mx = row[0];
    for (int o = 1; o < OUT_C; ++o) mx = fmaxf(mx, row[o]);
    float se = 0.f;
    for (int o = 0; o < OUT_C; ++o) se += expf(row[o] - mx);
    float lse = mx + logf(se);
    for (int o = 0; o < OUT_C; ++o) row[o] -= lse;
}

// ---------------------------------------------------------------- driver
static void run_gat_layer(const float* xin, int K, const float* W,
                          const float* a_src, const float* a_dst, const float* b,
                          const int* ei,
                          v16h* Ap, v16h* Bp,
                          float* h, float* as, float* ad, float* m, float* s,
                          float* ex, float* acc, float* t, hipStream_t stream) {
    const int T = 256;
    fill_kernel<<<(N_NODES * H_HEADS + T - 1) / T, T, 0, stream>>>(m, -INFINITY, N_NODES * H_HEADS);
    fill_kernel<<<(N_NODES * H_HEADS + T - 1) / T, T, 0, stream>>>(s, 0.0f, N_NODES * H_HEADS);
    fill_kernel<<<(N_NODES * C_CH  + T - 1) / T, T, 0, stream>>>(acc, 0.0f, N_NODES * C_CH);

    int na = (N_NODES / 16) * (K / 32) * 32;
    pack_a_kernel<<<(na + T - 1) / T, T, 0, stream>>>(xin, Ap, N_NODES, K);
    int nb = (K / 32) * (HC / 16) * 32;
    pack_b_kernel<<<(nb + T - 1) / T, T, 0, stream>>>(W, Bp, HC, K);

    int waves = (N_NODES / 16) * (HC / 64);   // 16x64 strip per wave
    gemm_wmma_packed<<<(waves + 7) / 8, 256, 0, stream>>>(Ap, Bp, h, N_NODES, HC, K);

    attn_coef_kernel<<<(N_NODES * H_HEADS + T - 1) / T, T, 0, stream>>>(h, a_src, a_dst, as, ad);

    int ne = EPRIME * H_HEADS;
    edge_max_kernel<<<(ne + T - 1) / T, T, 0, stream>>>(ei, as, ad, m);
    edge_exp_kernel<<<(ne + T - 1) / T, T, 0, stream>>>(ei, as, ad, m, ex, s);

    int nm = EPRIME * 32;
    edge_msg_kernel<<<(nm + T - 1) / T, T, 0, stream>>>(ei, ex, s, h, acc);

    bias_elu_kernel<<<(N_NODES * C_CH + T - 1) / T, T, 0, stream>>>(acc, b, t, N_NODES * C_CH);
}

extern "C" void kernel_launch(void* const* d_in, const int* in_sizes, int n_in,
                              void* d_out, int out_size, void* d_ws, size_t ws_size,
                              hipStream_t stream) {
    (void)in_sizes; (void)n_in; (void)out_size; (void)ws_size;
    const float* x   = (const float*)d_in[0];
    const int*   ei  = (const int*)  d_in[1];
    const float* W1  = (const float*)d_in[2];
    const float* av1 = (const float*)d_in[3];
    const float* ad1 = (const float*)d_in[4];
    const float* b1  = (const float*)d_in[5];
    const float* W2  = (const float*)d_in[6];
    const float* av2 = (const float*)d_in[7];
    const float* ad2 = (const float*)d_in[8];
    const float* b2  = (const float*)d_in[9];
    const float* Wo  = (const float*)d_in[10];
    const float* bo  = (const float*)d_in[11];

    char* ws = (char*)d_ws;
    size_t off = 0;
    auto alloc = [&](size_t bytes) -> void* {
        void* p = (void*)(ws + off);
        off += (bytes + 255) & ~(size_t)255;
        return p;
    };
    float* h   = (float*)alloc((size_t)N_NODES * HC      * sizeof(float)); // 51.2 MB
    float* acc = (float*)alloc((size_t)N_NODES * C_CH    * sizeof(float)); //  6.4 MB
    float* as  = (float*)alloc((size_t)N_NODES * H_HEADS * sizeof(float));
    float* ad  = (float*)alloc((size_t)N_NODES * H_HEADS * sizeof(float));
    float* m   = (float*)alloc((size_t)N_NODES * H_HEADS * sizeof(float));
    float* s   = (float*)alloc((size_t)N_NODES * H_HEADS * sizeof(float));
    float* ex  = (float*)alloc((size_t)EPRIME  * H_HEADS * sizeof(float)); // 27.2 MB
    float* t1  = (float*)alloc((size_t)N_NODES * C_CH    * sizeof(float));
    float* t2  = (float*)alloc((size_t)N_NODES * C_CH    * sizeof(float));
    v16h*  Ap  = (v16h*)alloc((size_t)(N_NODES / 16) * (IN_F / 32) * 32 * sizeof(v16h)); // 25.6 MB
    v16h*  Bp  = (v16h*)alloc((size_t)(IN_F / 32) * (HC / 16) * 32 * sizeof(v16h));      // 128 KB

    run_gat_layer(x,  IN_F, W1, av1, ad1, b1, ei, Ap, Bp, h, as, ad, m, s, ex, acc, t1, stream);
    run_gat_layer(t1, C_CH, W2, av2, ad2, b2, ei, Ap, Bp, h, as, ad, m, s, ex, acc, t2, stream);

    const int T = 256;
    out_proj_kernel<<<(N_NODES * OUT_C + T - 1) / T, T, 0, stream>>>(t2, Wo, bo, (float*)d_out);
    log_softmax_kernel<<<(N_NODES + T - 1) / T, T, 0, stream>>>((float*)d_out);
}
